// MHSAExtBiMambaLayer_47639777247252
// MI455X (gfx1250) — compile-verified, matrix operand order checked by reading, not run on hardware
//
#include <hip/hip_runtime.h>
#include <hip/hip_bf16.h>
#include <math.h>

// ---------------------------------------------------------------------------
// Types for CDNA5 WMMA
// ---------------------------------------------------------------------------
typedef __attribute__((ext_vector_type(16))) __bf16 v16bf;
typedef __attribute__((ext_vector_type(8)))  __bf16 v8bf;
typedef __attribute__((ext_vector_type(8)))  float  v8f;
typedef __attribute__((ext_vector_type(2)))  float  v2f;

__device__ inline v8f v8f_zero() {
  v8f z = {0.f, 0.f, 0.f, 0.f, 0.f, 0.f, 0.f, 0.f};
  return z;
}
__device__ inline float silu_f(float x) { return x / (1.f + __expf(-x)); }

// ---------------------------------------------------------------------------
// Dimensions (compile-time for this problem instance)
// ---------------------------------------------------------------------------
#define BATCH   16
#define SEQ     2048
#define DMODEL  512
#define DINNER  1024
#define DSTATE  16
#define DTRANK  32
#define MROWS   (BATCH * SEQ)          // 32768 flattened rows

// ---------------------------------------------------------------------------
// Tiled GEMM:  C[M,N] = A[M,K] @ W[N,K]^T (+ bias)
// bf16 WMMA (f32 accumulate); f32 -> bf16 conversion fused into LDS staging.
// Tile 128x128x32, 256 threads = 8 waves, each wave: 16-row strip x 128 cols.
// Software pipelined: double-buffered LDS, next k-tile staged into registers
// while WMMAs run; all 8 B fragments preloaded so the 8 WMMAs issue
// back-to-back (no per-WMMA wait-to-zero on DScnt).
// ---------------------------------------------------------------------------
#define LDSS 40   // LDS row stride in halves (keeps 16B alignment, skews banks)

__global__ __launch_bounds__(256) void k_gemm_bias(
    const float* __restrict__ A, int lda,
    const float* __restrict__ W, int ldw,
    const float* __restrict__ bias,
    float* __restrict__ C, int ldc,
    int M, int N, int K)
{
  __shared__ __bf16 sA[2][128 * LDSS];
  __shared__ __bf16 sB[2][128 * LDSS];

  const int tid    = threadIdx.x;
  const int lane   = tid & 31;
  const int wv     = tid >> 5;      // 0..7
  const int laneLo = lane & 15;
  const int hi     = lane >> 4;     // 0/1 half-wave
  const int gM     = blockIdx.y * 128;
  const int gN     = blockIdx.x * 128;

  v8f acc[8];
  #pragma unroll
  for (int i = 0; i < 8; ++i) acc[i] = v8f_zero();

  float4 ra[4], rb[4];

  // Stage next k-tile from global into registers (loads issue, no LDS yet).
  auto load_tile = [&](int k0) {
    #pragma unroll
    for (int i = 0; i < 4; ++i) {
      int idx = tid + i * 256;
      int r = idx >> 3, c4 = (idx & 7) * 4;
      ra[i] = *(const float4*)&A[(size_t)(gM + r) * lda + k0 + c4];
      int n = gN + r;
      rb[i] = make_float4(0.f, 0.f, 0.f, 0.f);
      if (n < N) rb[i] = *(const float4*)&W[(size_t)n * ldw + k0 + c4];
    }
  };
  // Convert f32 -> bf16 and commit registers to LDS buffer `cur`.
  auto store_tile = [&](int cur) {
    #pragma unroll
    for (int i = 0; i < 4; ++i) {
      int idx = tid + i * 256;
      int r = idx >> 3, c4 = (idx & 7) * 4;
      __bf16* da = &sA[cur][r * LDSS + c4];
      da[0] = (__bf16)ra[i].x; da[1] = (__bf16)ra[i].y;
      da[2] = (__bf16)ra[i].z; da[3] = (__bf16)ra[i].w;
      __bf16* db = &sB[cur][r * LDSS + c4];
      db[0] = (__bf16)rb[i].x; db[1] = (__bf16)rb[i].y;
      db[2] = (__bf16)rb[i].z; db[3] = (__bf16)rb[i].w;
    }
  };

  const int nk = K >> 5;
  load_tile(0);
  int cur = 0;

  for (int kt = 0; kt < nk; ++kt) {
    store_tile(cur);
    __syncthreads();
    if (kt + 1 < nk) load_tile((kt + 1) << 5);   // overlap HBM with WMMA below

    // A fragment: 16x32 bf16 per ISA layout. Lane (laneLo,hi) holds row
    // m=laneLo, k in {hi*8..hi*8+7} and {16+hi*8..}: two 16B chunks.
    union frag_t { v16bf v; v8bf h[2]; };
    frag_t afr;
    const int arow = (wv * 16 + laneLo) * LDSS;
    afr.h[0] = *(const v8bf*)&sA[cur][arow + hi * 8];
    afr.h[1] = *(const v8bf*)&sA[cur][arow + 16 + hi * 8];

    // Preload all 8 B fragments, then issue the 8 WMMAs back-to-back.
    frag_t bfr[8];
    #pragma unroll
    for (int nt = 0; nt < 8; ++nt) {
      const int brow = (nt * 16 + laneLo) * LDSS;
      bfr[nt].h[0] = *(const v8bf*)&sB[cur][brow + hi * 8];
      bfr[nt].h[1] = *(const v8bf*)&sB[cur][brow + 16 + hi * 8];
    }
    #pragma unroll
    for (int nt = 0; nt < 8; ++nt) {
      acc[nt] = __builtin_amdgcn_wmma_f32_16x16x32_bf16(
          false, afr.v, false, bfr[nt].v, (short)0, acc[nt], false, false);
    }
    cur ^= 1;
  }

  // Epilogue: accumulator VGPR r -> row r + hi*8 of the 16x16 tile.
  #pragma unroll
  for (int nt = 0; nt < 8; ++nt) {
    #pragma unroll
    for (int r = 0; r < 8; ++r) {
      int row = gM + wv * 16 + r + hi * 8;
      int col = gN + nt * 16 + laneLo;
      if (col < N) {
        float v = acc[nt][r];
        if (bias) v += bias[col];
        C[(size_t)row * ldc + col] = v;
      }
    }
  }
}

// ---------------------------------------------------------------------------
// Attention: seq axis is 16 (!), n=2048 is batch-like. One (n,h) pair per
// wave: scores 16x16 via fp32 WMMA 16x16x4 over hd=64, softmax over t,
// then P(16x16) @ V(16x64) via fp32 WMMA.
// ---------------------------------------------------------------------------
__global__ __launch_bounds__(128) void k_attn(
    const float* __restrict__ qkv,   // [16*2048][1536]
    float* __restrict__ o)           // [16*2048][512]
{
  __shared__ float sQ[4][16][64];
  __shared__ float sK[4][16][64];
  __shared__ float sV[4][16][64];
  __shared__ float sP[4][16][16];

  const int tid    = threadIdx.x;
  const int wv     = tid >> 5;            // 0..3
  const int lane   = tid & 31;
  const int laneLo = lane & 15;
  const int hi     = lane >> 4;
  const int pair   = blockIdx.x * 4 + wv; // 0..16383
  const int n      = pair >> 3;
  const int h      = pair & 7;

  // Stage q,k,v 16x64 tiles (float4 loads).
  #pragma unroll
  for (int i = 0; i < 8; ++i) {
    int f = lane + i * 32;                // 0..255 float4 slots
    int s = f >> 4, c4 = (f & 15) * 4;
    size_t base = ((size_t)s * SEQ + n) * 1536 + h * 64 + c4;
    *(float4*)&sQ[wv][s][c4] = *(const float4*)&qkv[base];
    *(float4*)&sK[wv][s][c4] = *(const float4*)&qkv[base + 512];
    *(float4*)&sV[wv][s][c4] = *(const float4*)&qkv[base + 1024];
  }
  __syncthreads();

  // scores[s][t] = sum_d q[s][d] * k[t][d], K-dim = 64 -> 16 wmma x4 steps.
  v8f sc = v8f_zero();
  #pragma unroll
  for (int kk = 0; kk < 16; ++kk) {
    int kk0 = kk * 4 + hi * 2;            // A 16x4 f32: VGPR0 K=0/K=2 split
    v2f a, b;
    a[0] = sQ[wv][laneLo][kk0];     a[1] = sQ[wv][laneLo][kk0 + 1];
    b[0] = sK[wv][laneLo][kk0];     b[1] = sK[wv][laneLo][kk0 + 1];
    sc = __builtin_amdgcn_wmma_f32_16x16x4_f32(
        false, a, false, b, (short)0, sc, false, false);
  }
  #pragma unroll
  for (int r = 0; r < 8; ++r)
    sP[wv][r + hi * 8][laneLo] = sc[r] * 0.125f;   // 1/sqrt(64)
  __syncthreads();

  // Softmax over t: one lane per row (lanes 0..15).
  if (lane < 16) {
    float m = -3.4e38f;
    #pragma unroll
    for (int t = 0; t < 16; ++t) m = fmaxf(m, sP[wv][lane][t]);
    float e[16], sum = 0.f;
    #pragma unroll
    for (int t = 0; t < 16; ++t) { e[t] = __expf(sP[wv][lane][t] - m); sum += e[t]; }
    float inv = 1.f / sum;
    #pragma unroll
    for (int t = 0; t < 16; ++t) sP[wv][lane][t] = e[t] * inv;
  }
  __syncthreads();

  // O = P @ V : 4 output tiles of 16x16, K=16 -> 4 wmma steps each.
  #pragma unroll
  for (int nt = 0; nt < 4; ++nt) {
    v8f ov = v8f_zero();
    #pragma unroll
    for (int kk = 0; kk < 4; ++kk) {
      int kk0 = kk * 4 + hi * 2;
      v2f a, b;
      a[0] = sP[wv][laneLo][kk0];         a[1] = sP[wv][laneLo][kk0 + 1];
      b[0] = sV[wv][kk0][nt * 16 + laneLo];
      b[1] = sV[wv][kk0 + 1][nt * 16 + laneLo];
      ov = __builtin_amdgcn_wmma_f32_16x16x4_f32(
          false, a, false, b, (short)0, ov, false, false);
    }
    #pragma unroll
    for (int r = 0; r < 8; ++r) {
      int s = r + hi * 8;
      o[((size_t)s * SEQ + n) * DMODEL + h * 64 + nt * 16 + laneLo] = ov[r];
    }
  }
}

// ---------------------------------------------------------------------------
// LayerNorm over last dim 512 with fused residual add: out = LN(res + add)
// ---------------------------------------------------------------------------
__global__ __launch_bounds__(128) void k_ln_res(
    const float* __restrict__ res, const float* __restrict__ add,
    const float* __restrict__ w, const float* __restrict__ b,
    float* __restrict__ out)
{
  __shared__ float buf[128];
  const size_t row = blockIdx.x;
  const int tid = threadIdx.x;

  float v[4];
  float s = 0.f;
  #pragma unroll
  for (int i = 0; i < 4; ++i) {
    int c = tid + i * 128;
    float t = res[row * DMODEL + c];
    if (add) t += add[row * DMODEL + c];
    v[i] = t; s += t;
  }
  buf[tid] = s; __syncthreads();
  for (int off = 64; off > 0; off >>= 1) {
    if (tid < off) buf[tid] += buf[tid + off];
    __syncthreads();
  }
  const float mu = buf[0] * (1.f / DMODEL);
  __syncthreads();

  float vs = 0.f;
  #pragma unroll
  for (int i = 0; i < 4; ++i) { float d = v[i] - mu; vs += d * d; }
  buf[tid] = vs; __syncthreads();
  for (int off = 64; off > 0; off >>= 1) {
    if (tid < off) buf[tid] += buf[tid + off];
    __syncthreads();
  }
  const float rstd = rsqrtf(buf[0] * (1.f / DMODEL) + 1e-5f);

  #pragma unroll
  for (int i = 0; i < 4; ++i) {
    int c = tid + i * 128;
    out[row * DMODEL + c] = (v[i] - mu) * rstd * w[c] + b[c];
  }
}

// ---------------------------------------------------------------------------
// Depthwise causal conv1d (width 4) + SiLU. u = first DINNER cols of xz.
// ---------------------------------------------------------------------------
__global__ __launch_bounds__(256) void k_conv_silu(
    const float* __restrict__ xz,      // [M][2048]
    const float* __restrict__ cw,      // [1024][1][4]
    const float* __restrict__ cb,      // [1024]
    float* __restrict__ u)             // [M][1024]
{
  size_t idx = (size_t)blockIdx.x * 256 + threadIdx.x; // over M*1024
  int d = (int)(idx & (DINNER - 1));
  size_t bl = idx >> 10;
  int l = (int)(bl & (SEQ - 1));
  size_t bb = bl >> 11;
  float acc = cb[d];
  #pragma unroll
  for (int j = 0; j < 4; ++j) {
    int ls = l - 3 + j;
    if (ls >= 0)
      acc += cw[d * 4 + j] * xz[((size_t)bb * SEQ + ls) * (2 * DINNER) + d];
  }
  u[idx] = silu_f(acc);
}

// ---------------------------------------------------------------------------
// Selective scan, fused: softplus(dtp) -> SSM recurrence (16 states) ->
// + u*D -> * silu(z). One thread per (b,d) channel, L=2048 sequential.
// B_t / C_t staged in LDS (broadcast to 256 channels of same batch).
// ---------------------------------------------------------------------------
__global__ __launch_bounds__(256) void k_scan(
    const float* __restrict__ xdbl,    // [M][64]  (dt|B|C)
    const float* __restrict__ dtp,     // [M][1024] pre-softplus
    const float* __restrict__ u,       // [M][1024]
    const float* __restrict__ xz,      // [M][2048] (z = cols 1024..2047)
    const float* __restrict__ Alog,    // [1024][16]
    const float* __restrict__ Dp,      // [1024]
    float* __restrict__ y)             // [M][1024]
{
  __shared__ float sB[DSTATE];
  __shared__ float sC[DSTATE];
  const int b = blockIdx.x >> 2;          // 16 batches x 4 channel groups
  const int g = blockIdx.x & 3;
  const int d = g * 256 + threadIdx.x;

  float A[DSTATE], h[DSTATE];
  #pragma unroll
  for (int n = 0; n < DSTATE; ++n) {
    A[n] = -__expf(Alog[d * DSTATE + n]);
    h[n] = 0.f;
  }
  const float Dd = Dp[d];

  for (int l = 0; l < SEQ; ++l) {
    size_t row = (size_t)b * SEQ + l;
    __syncthreads();
    if (threadIdx.x < 32) {
      float t = xdbl[row * 64 + 32 + threadIdx.x];
      if (threadIdx.x < 16) sB[threadIdx.x] = t;
      else                  sC[threadIdx.x - 16] = t;
    }
    __syncthreads();

    // Prefetch next row's channel data (CDNA5 global_prefetch_b8).
    if (l + 1 < SEQ)
      __builtin_prefetch(&dtp[(row + 1) * DINNER + d], 0, 0);

    float dpre  = dtp[row * DINNER + d];
    float delta = (dpre > 20.f) ? dpre : log1pf(__expf(dpre));
    float uu    = u[row * DINNER + d];
    float du    = delta * uu;
    float yv    = 0.f;
    #pragma unroll
    for (int n = 0; n < DSTATE; ++n) {
      float dA = __expf(delta * A[n]);
      h[n] = dA * h[n] + du * sB[n];
      yv  += h[n] * sC[n];
    }
    yv += uu * Dd;
    float z = xz[row * (2 * DINNER) + DINNER + d];
    yv *= silu_f(z);
    y[row * DINNER + d] = yv;
  }
}

// ---------------------------------------------------------------------------
// Flip along L (for backward mamba input).
// ---------------------------------------------------------------------------
__global__ __launch_bounds__(256) void k_flip512(
    const float* __restrict__ in, float* __restrict__ out)
{
  size_t idx = (size_t)blockIdx.x * 256 + threadIdx.x;  // M*512
  int c = (int)(idx & (DMODEL - 1));
  size_t rowf = idx >> 9;
  int l = (int)(rowf & (SEQ - 1));
  size_t bb = rowf >> 11;
  out[idx] = in[((bb * SEQ) + (SEQ - 1 - l)) * DMODEL + c];
}

// ---------------------------------------------------------------------------
// Concat [fo | flip_L(bo_f)] -> [M][1024]
// ---------------------------------------------------------------------------
__global__ __launch_bounds__(256) void k_concat(
    const float* __restrict__ fo, const float* __restrict__ bo_f,
    float* __restrict__ cat)
{
  size_t idx = (size_t)blockIdx.x * 256 + threadIdx.x;  // M*1024
  int c = (int)(idx & 1023);
  size_t row = idx >> 10;
  int l = (int)(row & (SEQ - 1));
  size_t bb = row >> 11;
  cat[idx] = (c < DMODEL)
      ? fo[row * DMODEL + c]
      : bo_f[((bb * SEQ) + (SEQ - 1 - l)) * DMODEL + (c - DMODEL)];
}

// ---------------------------------------------------------------------------
// Orchestration
// ---------------------------------------------------------------------------
extern "C" void kernel_launch(void* const* d_in, const int* in_sizes, int n_in,
                              void* d_out, int out_size, void* d_ws, size_t ws_size,
                              hipStream_t stream) {
  (void)in_sizes; (void)n_in; (void)out_size; (void)ws_size;

  const float* x        = (const float*)d_in[0];
  const float* mha_w_in = (const float*)d_in[1];
  const float* mha_b_in = (const float*)d_in[2];
  const float* mha_w_out= (const float*)d_in[3];
  const float* mha_b_out= (const float*)d_in[4];
  const float* ln1_w    = (const float*)d_in[5];
  const float* ln1_b    = (const float*)d_in[6];
  const float* ln2_w    = (const float*)d_in[7];
  const float* ln2_b    = (const float*)d_in[8];
  const float* comb_w   = (const float*)d_in[27];
  const float* comb_b   = (const float*)d_in[28];
  float* out = (float*)d_out;

  const size_t M = MROWS;
  float* ws = (float*)d_ws;
  size_t off = 0;
  float* qkvxz = ws + off; off += M * 2048;   // qkv (phase 1) then xz (phase 2)
  float* o_buf = ws + off; off += M * 512;
  float* oproj = ws + off; off += M * 512;
  float* x1    = ws + off; off += M * 512;
  float* ucv   = ws + off; off += M * 1024;
  float* xdbl  = ws + off; off += M * 64;
  float* dtp   = ws + off; off += M * 1024;
  float* yb    = ws + off; off += M * 1024;
  float* fo    = ws + off; off += M * 512;
  float* bo    = ws + off; off += M * 512;
  float* xb    = ws + off; off += M * 512;
  float* cat   = ws + off; off += M * 1024;
  float* mo    = ws + off; off += M * 512;

  const dim3 blk256(256), blk128(128);
  const int mTiles = (int)(M / 128);          // 256

  // ---- MHA ----
  // qkv = x @ w_in^T + b_in   [M,1536]
  k_gemm_bias<<<dim3(1536/128, mTiles), blk256, 0, stream>>>(
      x, 512, mha_w_in, 512, mha_b_in, qkvxz, 1536, (int)M, 1536, 512);
  // attention (one (n,h) pair per wave)
  k_attn<<<dim3(16384/4), blk128, 0, stream>>>(qkvxz, o_buf);
  // out proj
  k_gemm_bias<<<dim3(512/128, mTiles), blk256, 0, stream>>>(
      o_buf, 512, mha_w_out, 512, mha_b_out, oproj, 512, (int)M, 512, 512);
  // x1 = LN(x + attn)
  k_ln_res<<<dim3((unsigned)M), blk128, 0, stream>>>(x, oproj, ln1_w, ln1_b, x1);

  // ---- Bi-directional Mamba ----
  k_flip512<<<dim3((unsigned)(M*512/256)), blk256, 0, stream>>>(x1, xb);

  for (int dir = 0; dir < 2; ++dir) {
    const float* in_w    = (const float*)d_in[9  + dir*9];
    const float* conv_w  = (const float*)d_in[10 + dir*9];
    const float* conv_b  = (const float*)d_in[11 + dir*9];
    const float* xproj_w = (const float*)d_in[12 + dir*9];
    const float* dt_w    = (const float*)d_in[13 + dir*9];
    const float* dt_b    = (const float*)d_in[14 + dir*9];
    const float* Alog    = (const float*)d_in[15 + dir*9];
    const float* Dp      = (const float*)d_in[16 + dir*9];
    const float* out_w   = (const float*)d_in[17 + dir*9];
    const float* xin     = (dir == 0) ? x1 : xb;
    float*       yout    = (dir == 0) ? fo : bo;

    // xz = xin @ in_w^T   [M,2048]
    k_gemm_bias<<<dim3(2048/128, mTiles), blk256, 0, stream>>>(
        xin, 512, in_w, 512, nullptr, qkvxz, 2048, (int)M, 2048, 512);
    // u = silu(conv1d(u))
    k_conv_silu<<<dim3((unsigned)(M*1024/256)), blk256, 0, stream>>>(
        qkvxz, conv_w, conv_b, ucv);
    // x_dbl = u @ xproj_w^T  [M,64]
    k_gemm_bias<<<dim3(1, mTiles), blk256, 0, stream>>>(
        ucv, 1024, xproj_w, 1024, nullptr, xdbl, 64, (int)M, 64, 1024);
    // dtp = x_dbl[:, :32] @ dt_w^T + dt_b  [M,1024]
    k_gemm_bias<<<dim3(1024/128, mTiles), blk256, 0, stream>>>(
        xdbl, 64, dt_w, 32, dt_b, dtp, 1024, (int)M, 1024, 32);
    // fused selective scan + gate
    k_scan<<<dim3(16*4), blk256, 0, stream>>>(
        xdbl, dtp, ucv, qkvxz, Alog, Dp, yb);
    // y @ out_w^T  [M,512]
    k_gemm_bias<<<dim3(512/128, mTiles), blk256, 0, stream>>>(
        yb, 1024, out_w, 1024, nullptr, yout, 512, (int)M, 512, 1024);
  }

  // ---- Combine + final LN ----
  k_concat<<<dim3((unsigned)(M*1024/256)), blk256, 0, stream>>>(fo, bo, cat);
  k_gemm_bias<<<dim3(512/128, mTiles), blk256, 0, stream>>>(
      cat, 1024, comb_w, 1024, comb_b, mo, 512, (int)M, 512, 1024);
  k_ln_res<<<dim3((unsigned)M), blk128, 0, stream>>>(x1, mo, ln2_w, ln2_b, out);
}
